// GCN_6983616824079
// MI455X (gfx1250) — matrix operand order repaired and checked by reading, not compiled, not run on hardware
//
#include <hip/hip_runtime.h>

typedef _Float16 v16h __attribute__((ext_vector_type(16)));
typedef _Float16 h8   __attribute__((ext_vector_type(8)));
typedef float    v8f  __attribute__((ext_vector_type(8)));

#define DH 128
#define LDSPITCH 136   // 128 + 8 halves pad -> 272B row stride, 16B aligned, bank-spread

// ---------------- utility kernels ----------------

__global__ void zero_f32(float* __restrict__ p, long long n) {
    long long i = (long long)blockIdx.x * blockDim.x + threadIdx.x;
    if (i < n) p[i] = 0.0f;
}

__global__ void fill_f32(float* __restrict__ p, float v, long long n) {
    long long i = (long long)blockIdx.x * blockDim.x + threadIdx.x;
    if (i < n) p[i] = v;
}

__global__ void degree_kernel(const long long* __restrict__ dst,
                              float* __restrict__ deg, int E) {
    int i = blockIdx.x * blockDim.x + threadIdx.x;
    if (i < E) atomicAdd(&deg[(int)dst[i]], 1.0f);
}

__global__ void rsqrt_kernel(float* __restrict__ deg, int n) {
    int i = blockIdx.x * blockDim.x + threadIdx.x;
    if (i < n) deg[i] = rsqrtf(deg[i]);
}

// ---------------- WMMA GEMM: Out[M,128] = A[M,128] @ W[128,128] ----------------
// f32 inputs converted to f16, f32 accumulate. One wave owns a 16-row tile and
// all 8 column tiles; W is staged transposed in LDS as f16.

__global__ void __launch_bounds__(128)
gemm_wmma(const float* __restrict__ A, const float* __restrict__ W,
          float* __restrict__ Out, int M) {
    __shared__ _Float16 sW[DH * LDSPITCH];   // sW[n*PITCH + k] = (f16) W[k*128 + n]

    const int tid = threadIdx.x;
    // cooperative transpose-stage of W (coalesced global reads: 128 floats/row)
    {
        const int n = tid;                   // 0..127
        for (int k = 0; k < DH; ++k)
            sW[n * LDSPITCH + k] = (_Float16)W[k * DH + n];
    }
    __syncthreads();

    const int lane = tid & 31;
    const int wave = tid >> 5;
    const int m0   = blockIdx.x * 64 + wave * 16;
    const int mrow = m0 + (lane & 15);
    const bool arow_ok = (mrow < M);

    v8f acc[8] = {};

    #pragma unroll
    for (int kt = 0; kt < 4; ++kt) {
        // ---- A fragment: 16x32 f16 (lane M = lane&15, K halves 0-7 / 16-23 by lane group)
        v16h a{};
        if (arow_ok) {
            const int koff = kt * 32 + ((lane >> 4) << 3);   // +0 or +8
            const float* ap = A + (size_t)mrow * DH + koff;
            const float4 f0 = *(const float4*)(ap);
            const float4 f1 = *(const float4*)(ap + 4);
            const float4 f2 = *(const float4*)(ap + 16);
            const float4 f3 = *(const float4*)(ap + 20);
            a[0]=(_Float16)f0.x; a[1]=(_Float16)f0.y; a[2]=(_Float16)f0.z; a[3]=(_Float16)f0.w;
            a[4]=(_Float16)f1.x; a[5]=(_Float16)f1.y; a[6]=(_Float16)f1.z; a[7]=(_Float16)f1.w;
            a[8]=(_Float16)f2.x; a[9]=(_Float16)f2.y; a[10]=(_Float16)f2.z; a[11]=(_Float16)f2.w;
            a[12]=(_Float16)f3.x; a[13]=(_Float16)f3.y; a[14]=(_Float16)f3.z; a[15]=(_Float16)f3.w;
        }

        const int kb = kt * 32 + ((lane >> 4) << 3);         // +0 or +8
        #pragma unroll
        for (int nt = 0; nt < 8; ++nt) {
            // ---- B fragment: 32x16 f16 (lane N = lane&15, K halves 0-7 / 16-23 by lane group)
            const int n = nt * 16 + (lane & 15);
            const h8 lo = *(const h8*)(&sW[n * LDSPITCH + kb]);
            const h8 hi = *(const h8*)(&sW[n * LDSPITCH + kb + 16]);
            v16h b;
            #pragma unroll
            for (int i = 0; i < 8; ++i) { b[i] = lo[i]; b[i + 8] = hi[i]; }

            acc[nt] = __builtin_amdgcn_wmma_f32_16x16x32_f16(
                false, a, false, b, (short)0, acc[nt], false, false);
        }
    }

    // ---- store D: lane holds rows (lane>=16 ? +8 : +0) + r, col = lane&15
    const int nb = lane & 15;
    const int rb = (lane >> 4) * 8;
    #pragma unroll
    for (int nt = 0; nt < 8; ++nt) {
        #pragma unroll
        for (int r = 0; r < 8; ++r) {
            const int row = m0 + rb + r;
            if (row < M) Out[(size_t)row * DH + nt * 16 + nb] = acc[nt][r];
        }
    }
}

// ---------------- edge scatter: agg[dst] += h[src] * dinv[src]*dinv[dst] ----------------
// one wave per edge (incl. self-loops at e >= E); lane handles a float4 of the 128 feats.

__global__ void __launch_bounds__(256)
scatter_edges(const float* __restrict__ h,
              const long long* __restrict__ src,
              const long long* __restrict__ dst,
              const float* __restrict__ dinv,
              float* __restrict__ agg, int E, int Nn) {
    const long long gid = (long long)blockIdx.x * blockDim.x + threadIdx.x;
    const long long e = gid >> 5;
    const int lane = (int)(gid & 31);
    if (e >= (long long)E + Nn) return;

    int s, d;
    if (e < E) { s = (int)src[e]; d = (int)dst[e]; }
    else       { s = d = (int)(e - E); }

    const float nrm = dinv[s] * dinv[d];
    const float4 v = *(const float4*)(h + (size_t)s * DH + lane * 4);
    float* o = agg + (size_t)d * DH + lane * 4;
    atomicAdd(o + 0, v.x * nrm);
    atomicAdd(o + 1, v.y * nrm);
    atomicAdd(o + 2, v.z * nrm);
    atomicAdd(o + 3, v.w * nrm);
}

__global__ void bias_relu_kernel(float* __restrict__ agg, const float* __restrict__ b,
                                 long long total) {
    long long i = (long long)blockIdx.x * blockDim.x + threadIdx.x;
    if (i >= total) return;
    float v = agg[i] + b[(int)(i & (DH - 1))];
    agg[i] = v > 0.0f ? v : 0.0f;
}

// ---------------- pooling (batch is sorted -> run-length local accumulation) ----------------

__global__ void pool_kernel(const float* __restrict__ agg2, const float* __restrict__ b2,
                            const long long* __restrict__ batch,
                            float* __restrict__ pooled, int Nn) {
    const long long gid = (long long)blockIdx.x * blockDim.x + threadIdx.x;
    const int col = (int)(gid & (DH - 1));
    const int n0  = (int)(gid >> 7) * 32;
    if (n0 >= Nn) return;
    const int nend = min(n0 + 32, Nn);
    const float bc = b2[col];
    long long curg = batch[n0];
    float a = 0.0f;
    for (int n = n0; n < nend; ++n) {
        long long g = batch[n];
        if (g != curg) { atomicAdd(&pooled[curg * DH + col], a); a = 0.0f; curg = g; }
        a += agg2[(size_t)n * DH + col] + bc;
    }
    atomicAdd(&pooled[curg * DH + col], a);
}

__global__ void counts_kernel(const long long* __restrict__ batch,
                              float* __restrict__ counts, int Nn) {
    const int c = blockIdx.x * blockDim.x + threadIdx.x;
    const int n0 = c * 32;
    if (n0 >= Nn) return;
    const int nend = min(n0 + 32, Nn);
    long long curg = batch[n0];
    float a = 0.0f;
    for (int n = n0; n < nend; ++n) {
        long long g = batch[n];
        if (g != curg) { atomicAdd(&counts[curg], a); a = 0.0f; curg = g; }
        a += 1.0f;
    }
    atomicAdd(&counts[curg], a);
}

__global__ void final_kernel(const float* __restrict__ pooled, const float* __restrict__ counts,
                             const float* __restrict__ Wfc, const float* __restrict__ bfc,
                             float* __restrict__ out, int G) {
    const int t = threadIdx.x;
    if (t >= G * 2) return;
    const int g = t >> 1, c = t & 1;
    float s = 0.0f;
    for (int k = 0; k < DH; ++k) s += pooled[g * DH + k] * Wfc[k * 2 + c];
    float cnt = counts[g];
    cnt = cnt > 1.0f ? cnt : 1.0f;
    out[g * 2 + c] = s / cnt + bfc[c];
}

// ---------------- host launch ----------------

extern "C" void kernel_launch(void* const* d_in, const int* in_sizes, int n_in,
                              void* d_out, int out_size, void* d_ws, size_t ws_size,
                              hipStream_t stream) {
    const float*     x     = (const float*)d_in[0];
    const long long* eidx  = (const long long*)d_in[1];
    const long long* batch = (const long long*)d_in[2];
    const float*     W1    = (const float*)d_in[3];
    const float*     b1    = (const float*)d_in[4];
    const float*     W2    = (const float*)d_in[5];
    const float*     b2    = (const float*)d_in[6];
    const float*     Wfc   = (const float*)d_in[7];
    const float*     bfc   = (const float*)d_in[8];
    float*           out   = (float*)d_out;

    const int Nn = in_sizes[0] / DH;     // 50000
    const int E  = in_sizes[1] / 2;      // 800000
    const int G  = out_size / 2;         // 64 graphs
    const long long* srcp = eidx;
    const long long* dstp = eidx + E;

    // workspace layout (floats): deg[Nn] | hx[Nn*128] | agg[Nn*128] | pooled[G*128] | counts[G]
    float* ws     = (float*)d_ws;
    float* deg    = ws;
    float* hx     = deg + Nn;
    float* agg    = hx + (size_t)Nn * DH;
    float* pooled = agg + (size_t)Nn * DH;
    float* counts = pooled + (size_t)G * DH;
    (void)ws_size; (void)n_in;

    const long long nd    = (long long)Nn * DH;
    const long long ztail = nd + (long long)G * DH + G;   // agg..counts contiguous

    auto blk = [](long long n, int t) { return (unsigned)((n + t - 1) / t); };

    // 1) deg = 1.0 (self-loop), zero agg/pooled/counts
    fill_f32<<<blk(Nn, 256), 256, 0, stream>>>(deg, 1.0f, Nn);
    zero_f32<<<blk(ztail, 256), 256, 0, stream>>>(agg, ztail);
    // 2) degree accumulation + D^-1/2
    degree_kernel<<<blk(E, 256), 256, 0, stream>>>(dstp, deg, E);
    rsqrt_kernel<<<blk(Nn, 256), 256, 0, stream>>>(deg, Nn);
    // 3) layer 1: hx = x @ W1  (WMMA)
    gemm_wmma<<<blk(Nn, 64), 128, 0, stream>>>(x, W1, hx, Nn);
    // 4) agg = scatter(hx)
    {
        const long long waves = (long long)E + Nn;
        scatter_edges<<<blk(waves * 32, 256), 256, 0, stream>>>(hx, srcp, dstp, deg, agg, E, Nn);
    }
    // 5) h1 = relu(agg + b1) in place
    bias_relu_kernel<<<blk(nd, 256), 256, 0, stream>>>(agg, b1, nd);
    // 6) layer 2: hx = h1 @ W2  (WMMA)
    gemm_wmma<<<blk(Nn, 64), 128, 0, stream>>>(agg, W2, hx, Nn);
    // 7) re-zero agg, scatter layer 2
    zero_f32<<<blk(nd, 256), 256, 0, stream>>>(agg, nd);
    {
        const long long waves = (long long)E + Nn;
        scatter_edges<<<blk(waves * 32, 256), 256, 0, stream>>>(hx, srcp, dstp, deg, agg, E, Nn);
    }
    // 8) pooled sums (b2 folded in), counts, classifier (mean folded in)
    pool_kernel<<<blk((long long)((Nn + 31) / 32) * DH, 256), 256, 0, stream>>>(agg, b2, batch, pooled, Nn);
    counts_kernel<<<blk((Nn + 31) / 32, 256), 256, 0, stream>>>(batch, counts, Nn);
    final_kernel<<<1, 128, 0, stream>>>(pooled, counts, Wfc, bfc, out, G);
}
